// SlidingAttention_70471823392815
// MI455X (gfx1250) — matrix operand order repaired and checked by reading, not compiled, not run on hardware
//
#include <hip/hip_runtime.h>
#include <hip/hip_bf16.h>

// ---------------- constants ----------------
#define SEQ       2048
#define HIDDEN    4096
#define HEAD_DIM  128
#define NQH       32
#define NKVH      8
#define WINDOW    1024

typedef __attribute__((ext_vector_type(16))) __bf16 v16bf;
typedef __attribute__((ext_vector_type(8)))  __bf16 v8bf;
typedef __attribute__((ext_vector_type(8)))  float  v8f;
typedef int          v4i   __attribute__((vector_size(16)));
typedef unsigned int u32x4 __attribute__((vector_size(16)));
typedef int          i32x4 __attribute__((vector_size(16)));
typedef int          i32x8 __attribute__((vector_size(32)));

#define GLOBAL_AS __attribute__((address_space(1)))
#define LDS_AS    __attribute__((address_space(3)))

// ---- CDNA5 async global->LDS copies (ASYNCcnt), guarded by __has_builtin ----
#if defined(__has_builtin)
#if __has_builtin(__builtin_amdgcn_global_load_async_to_lds_b128)
#define HAVE_ASYNC_LDS 1
#endif
#if __has_builtin(__builtin_amdgcn_tensor_load_to_lds)
#define HAVE_TDM 1
#endif
#endif

#ifdef HAVE_ASYNC_LDS
__device__ __forceinline__ void async_copy_b128(const __bf16* g, __bf16* l) {
    __builtin_amdgcn_global_load_async_to_lds_b128(
        (GLOBAL_AS v4i*)g, (LDS_AS v4i*)l, 0, 0);
}
#if __has_builtin(__builtin_amdgcn_s_wait_asynccnt)
#define WAIT_ASYNC() __builtin_amdgcn_s_wait_asynccnt(0)
#else
#define WAIT_ASYNC() asm volatile("s_wait_asynccnt 0" ::: "memory")
#endif
#endif

#ifdef HAVE_TDM
// Tensor Data Mover: 2D tile load, global[rows x 128 bf16, stride 1024 elems]
// -> LDS rows padded to 136 elems (pad 4 DWORDs every 64 DWORDs).
__device__ __forceinline__ void tdm_load_k_tile(const __bf16* gbase, void* lds_base,
                                                int rows) {
    unsigned long long ga = (unsigned long long)gbase;
    unsigned lds = (unsigned)(unsigned long long)(LDS_AS void*)lds_base;
    u32x4 g0 = {
        1u,                                    // count=1 valid, no gather
        lds,                                   // lds_addr (bytes)
        (unsigned)(ga & 0xFFFFFFFFu),          // global_addr[31:0]
        (unsigned)((ga >> 32) & 0x1FFFFFFu) | 0x80000000u  // addr[56:32] | type=2
    };
    i32x8 g1 = {
        (int)((1u << 16) | (1u << 20) | (5u << 22) | (3u << 25)),
        // ^ data_size=2B, pad_enable, pad_interval=64dw, pad_amount=4dw
        (int)(128u << 16),                     // tensor_dim0 = 128
        (int)((unsigned)SEQ << 16),            // tensor_dim1 = 2048
        (int)(128u << 16),                     // tile_dim0 = 128
        rows,                                  // tile_dim1 (tile_dim2 = 0)
        1024,                                  // tensor_dim0_stride = 1024 elems
        0, 0
    };
    i32x4 z4 = {0, 0, 0, 0};
#if __clang_major__ >= 23
    i32x8 z8 = {0, 0, 0, 0, 0, 0, 0, 0};
    __builtin_amdgcn_tensor_load_to_lds(g0, g1, z4, z4, z8, 0);
#else
    __builtin_amdgcn_tensor_load_to_lds(g0, g1, z4, z4, 0);
#endif
}
#if __has_builtin(__builtin_amdgcn_s_wait_tensorcnt)
#define WAIT_TENSOR() __builtin_amdgcn_s_wait_tensorcnt(0)
#else
#define WAIT_TENSOR() asm volatile("s_wait_tensorcnt 0" ::: "memory")
#endif
#endif

__device__ __forceinline__ __bf16 f2bf(float x) {
    unsigned u = __builtin_bit_cast(unsigned, x);
    unsigned r = (u + 0x7FFFu + ((u >> 16) & 1u)) >> 16;
    unsigned short h = (unsigned short)r;
    return __builtin_bit_cast(__bf16, h);
}

__device__ __forceinline__ v8f v8f_zero() {
    v8f z = {0.f, 0.f, 0.f, 0.f, 0.f, 0.f, 0.f, 0.f};
    return z;
}

// ---------------- f32 -> bf16 conversion ----------------
__global__ __launch_bounds__(256)
void convert_f32_bf16(const float* __restrict__ in, __bf16* __restrict__ out, int n) {
    int i = blockIdx.x * 256 + threadIdx.x;
    if (i < n) out[i] = f2bf(in[i]);
}

// ---------------- tiled bf16 WMMA GEMM: C[M,N] = A[M,K] @ B[K,N] ----------------
// block: 128x128 tile, 256 threads = 8 waves in 2(M) x 4(N); wave tile 64x32.
template<bool OUT_BF16>
__global__ __launch_bounds__(256)
void gemm_bf16_wmma(const __bf16* __restrict__ A, const __bf16* __restrict__ B,
                    float* __restrict__ Cf, __bf16* __restrict__ Cb,
                    int M, int N, int K)
{
    __shared__ __align__(16) __bf16 As[128][40];  // [m][k]
    __shared__ __align__(16) __bf16 Bt[128][40];  // [n][k]  (B staged transposed)

    const int tid   = threadIdx.x;
    const int lane  = tid & 31;
    const int wid   = tid >> 5;
    const int waveM = wid >> 2;       // 0..1
    const int waveN = wid & 3;        // 0..3
    const int bm    = blockIdx.y * 128;
    const int bn    = blockIdx.x * 128;
    const int r     = lane & 15;
    const int hf    = lane >> 4;

    v8f acc[4][2];
#pragma unroll
    for (int mt = 0; mt < 4; ++mt)
#pragma unroll
        for (int nt = 0; nt < 2; ++nt) acc[mt][nt] = v8f_zero();

    for (int k0 = 0; k0 < K; k0 += 32) {
        // ---- stage A tile 128x32 (8KB): async global->LDS when available ----
#ifdef HAVE_ASYNC_LDS
        {
#pragma unroll
            for (int it = 0; it < 2; ++it) {
                int tb   = (tid + it * 256) * 16;   // byte index within 128x64B tile
                int rowA = tb >> 6;                 // 64 bytes per row
                int colb = tb & 63;
                const __bf16* gp = A + (size_t)(bm + rowA) * K + k0 + (colb >> 1);
                async_copy_b128(gp, &As[rowA][colb >> 1]);
            }
        }
#else
        {
            int rowA = tid >> 1;
            int colA = (tid & 1) * 16;
            const __bf16* gp = A + (size_t)(bm + rowA) * K + k0 + colA;
            *(v8bf*)&As[rowA][colA]     = *(const v8bf*)(gp);
            *(v8bf*)&As[rowA][colA + 8] = *(const v8bf*)(gp + 8);
        }
#endif
        // ---- stage B tile 32x128 transposed (scatter, must be manual) ----
        {
            int rowB = tid >> 3;            // 0..31 (k)
            int nb   = (tid & 7) * 16;      // n base
            const __bf16* gp = B + (size_t)(k0 + rowB) * N + bn + nb;
            v8bf b0 = *(const v8bf*)(gp);
            v8bf b1 = *(const v8bf*)(gp + 8);
#pragma unroll
            for (int i = 0; i < 8; ++i) {
                Bt[nb + i][rowB]     = b0[i];
                Bt[nb + 8 + i][rowB] = b1[i];
            }
        }
        // prefetch next K-step tiles into cache (global_prefetch_b8)
        if (k0 + 32 < K) {
            __builtin_prefetch(A + (size_t)(bm + (tid >> 1)) * K + k0 + 32, 0, 3);
            __builtin_prefetch(B + (size_t)(k0 + 32 + (tid >> 3)) * N + bn + (tid & 7) * 16, 0, 3);
        }
#ifdef HAVE_ASYNC_LDS
        WAIT_ASYNC();
#endif
        __syncthreads();

        // B fragments (contiguous k order: lane half hf holds k 16hf..16hf+15, same as A)
        v16bf bfrag[2];
#pragma unroll
        for (int nt = 0; nt < 2; ++nt) {
            const __bf16* bp = &Bt[waveN * 32 + nt * 16 + r][hf * 16];
            v16bf b;
#pragma unroll
            for (int i = 0; i < 16; ++i) b[i] = bp[i];
            bfrag[nt] = b;
        }
#pragma unroll
        for (int mt = 0; mt < 4; ++mt) {
            // A fragment (lane half selects K 0-15 vs 16-31, matching B)
            const __bf16* ap = &As[waveM * 64 + mt * 16 + r][0];
            v16bf a;
#pragma unroll
            for (int i = 0; i < 8; ++i) {
                a[i]     = ap[hf * 16 + i];
                a[8 + i] = ap[hf * 16 + 8 + i];
            }
#pragma unroll
            for (int nt = 0; nt < 2; ++nt) {
                acc[mt][nt] = __builtin_amdgcn_wmma_f32_16x16x32_bf16(
                    false, a, false, bfrag[nt], (short)0, acc[mt][nt], false, false);
            }
        }
        __syncthreads();
    }

    // epilogue: C layout row = rr + 8*hf, col = lane&15
#pragma unroll
    for (int mt = 0; mt < 4; ++mt)
#pragma unroll
        for (int nt = 0; nt < 2; ++nt)
#pragma unroll
            for (int rr = 0; rr < 8; ++rr) {
                int row = bm + waveM * 64 + mt * 16 + rr + 8 * hf;
                int col = bn + waveN * 32 + nt * 16 + r;
                float v = acc[mt][nt][rr];
                if (OUT_BF16) Cb[(size_t)row * N + col] = f2bf(v);
                else          Cf[(size_t)row * N + col] = v;
            }
}

// ---------------- RoPE: f32 [S, nh, 128] -> bf16 rotated ----------------
__global__ __launch_bounds__(256)
void rope_to_bf16(const float* __restrict__ in, __bf16* __restrict__ out, int nheads) {
    int idx = blockIdx.x * 256 + threadIdx.x;
    int total = SEQ * nheads * 64;
    if (idx >= total) return;
    int d = idx & 63;
    int h = (idx >> 6) % nheads;
    int s = idx / (nheads * 64);
    const float* row = in + ((size_t)s * nheads + h) * HEAD_DIM;
    float a = row[d], b = row[d + 64];
    // inv_freq = theta^(-d/64) = exp(-d * ln(10000)/64)
    float invf = __expf(-(float)d * 0.14391156831212787f);
    float ang = (float)s * invf;
    float sn, cs;
    __sincosf(ang, &sn, &cs);
    __bf16* orow = out + ((size_t)s * nheads + h) * HEAD_DIM;
    orow[d]      = f2bf(a * cs - b * sn);
    orow[d + 64] = f2bf(a * sn + b * cs);
}

// ---------------- sliding-window flash attention ----------------
// grid: (SEQ/128, NQH). 256 threads = 8 waves; wave w handles q rows [qblk+16w, +16).
// K staged row-major in LDS via the Tensor Data Mover (TENSORcnt) when available
// (async / manual fallback); V staged transposed [d][key]; P bounced through LDS.
__global__ __launch_bounds__(256)
void attn_swa_wmma(const __bf16* __restrict__ Q, const __bf16* __restrict__ Kb,
                   const __bf16* __restrict__ Vb, __bf16* __restrict__ O)
{
    __shared__ __align__(16) __bf16 Ks[64][136];    // [key][d], 136 = 128 + 4dw pad
    __shared__ __align__(16) __bf16 Vt[128][72];    // [d][key]
    __shared__ __align__(16) __bf16 Ps[8][16][40];  // per-wave P scratch [m][k]

    const int tid  = threadIdx.x;
    const int lane = tid & 31;
    const int wid  = tid >> 5;
    const int r    = lane & 15;
    const int hf   = lane >> 4;
    const int h    = blockIdx.y;
    const int kvh  = h >> 2;
    const int qblk = blockIdx.x * 128;
    const int qb   = qblk + wid * 16;

    // Q fragments: 4 chunks of 16x32 over head dim, loaded straight from global
    v16bf qf[4];
#pragma unroll
    for (int c = 0; c < 4; ++c) {
        const __bf16* qp = Q + ((size_t)(qb + r) * NQH + h) * HEAD_DIM + c * 32;
        v16bf a;
#pragma unroll
        for (int i = 0; i < 8; ++i) {
            a[i]     = qp[hf * 16 + i];
            a[8 + i] = qp[hf * 16 + 8 + i];
        }
        qf[c] = a;
    }

    v8f o[8];
#pragma unroll
    for (int t = 0; t < 8; ++t) o[t] = v8f_zero();
    float mr[8], lr[8];
#pragma unroll
    for (int rr = 0; rr < 8; ++rr) { mr[rr] = -1.0e30f; lr[rr] = 0.0f; }

    const int wlo = (qb - (WINDOW - 1)) > 0 ? (qb - (WINDOW - 1)) : 0;
    const int whi = qb + 15;
    int c0lo = qblk - (WINDOW - 1);
    if (c0lo < 0) c0lo = 0;
    c0lo &= ~63;

    for (int c0 = c0lo; c0 <= qblk + 127; c0 += 64) {
        __syncthreads();
        // ---- stage K rows [c0, c0+64): TDM > async > manual ----
#if defined(HAVE_TDM)
        if (wid == 0) {
            const __bf16* gk = Kb + ((size_t)c0 * NKVH + kvh) * HEAD_DIM;
            tdm_load_k_tile(gk, &Ks[0][0], 64);
            WAIT_TENSOR();
        }
#elif defined(HAVE_ASYNC_LDS)
#pragma unroll
        for (int it = 0; it < 4; ++it) {
            int tb = (tid + it * 256) * 16;   // byte index within 64x256B tile
            int kr = tb >> 8;                 // 256 bytes per key row
            int cb = tb & 255;
            const __bf16* kp = Kb + ((size_t)(c0 + kr) * NKVH + kvh) * HEAD_DIM + (cb >> 1);
            async_copy_b128(kp, &Ks[kr][cb >> 1]);
        }
#endif
        // ---- stage V transposed (scatter, manual) + K manual fallback ----
#pragma unroll
        for (int it = 0; it < 4; ++it) {
            int idx = tid * 8 + it * 2048;
            int kr  = idx >> 7;     // 0..63
            int cc  = idx & 127;    // d, multiple of 8
#if !defined(HAVE_TDM) && !defined(HAVE_ASYNC_LDS)
            const __bf16* kp = Kb + ((size_t)(c0 + kr) * NKVH + kvh) * HEAD_DIM + cc;
            *(v8bf*)&Ks[kr][cc] = *(const v8bf*)kp;
#endif
            const __bf16* vp = Vb + ((size_t)(c0 + kr) * NKVH + kvh) * HEAD_DIM + cc;
            v8bf vv = *(const v8bf*)vp;
#pragma unroll
            for (int i = 0; i < 8; ++i) Vt[cc + i][kr] = vv[i];
        }
#if !defined(HAVE_TDM) && defined(HAVE_ASYNC_LDS)
        WAIT_ASYNC();
#endif
        __syncthreads();

        int kb0 = wlo & ~31;
        if (kb0 < c0) kb0 = c0;
        int kbe = (whi < c0 + 63) ? whi : c0 + 63;

        for (int kb = kb0; kb <= kbe; kb += 32) {
            // ---- scores for both 16-key subtiles ----
            v8f sc[2];
#pragma unroll
            for (int j = 0; j < 2; ++j) {
                v8f s = v8f_zero();
#pragma unroll
                for (int c = 0; c < 4; ++c) {
                    const __bf16* bp = &Ks[(kb - c0) + j * 16 + r][c * 32 + hf * 16];
                    v16bf b;
#pragma unroll
                    for (int i = 0; i < 16; ++i) b[i] = bp[i];
                    s = __builtin_amdgcn_wmma_f32_16x16x32_bf16(
                        false, qf[c], false, b, (short)0, s, false, false);
                }
                // scale + sliding-window causal mask
#pragma unroll
                for (int rr = 0; rr < 8; ++rr) {
                    int m  = qb + rr + 8 * hf;
                    int kk = kb + j * 16 + r;
                    float sv = s[rr] * 0.08838834764831845f;
                    if (kk > m || kk + WINDOW <= m) sv = -1.0e30f;
                    s[rr] = sv;
                }
                sc[j] = s;
            }
            // ---- online softmax (16-lane butterfly per half-wave) ----
            v8f tm;
#pragma unroll
            for (int rr = 0; rr < 8; ++rr) tm[rr] = fmaxf(sc[0][rr], sc[1][rr]);
#pragma unroll
            for (int off = 1; off < 16; off <<= 1)
#pragma unroll
                for (int rr = 0; rr < 8; ++rr)
                    tm[rr] = fmaxf(tm[rr], __shfl_xor(tm[rr], off, 32));

            float corr[8];
            v8f p0, p1;
#pragma unroll
            for (int rr = 0; rr < 8; ++rr) {
                float nm = fmaxf(mr[rr], tm[rr]);
                corr[rr] = __expf(mr[rr] - nm);
                mr[rr]   = nm;
                p0[rr]   = __expf(sc[0][rr] - nm);
                p1[rr]   = __expf(sc[1][rr] - nm);
            }
            v8f ps;
#pragma unroll
            for (int rr = 0; rr < 8; ++rr) ps[rr] = p0[rr] + p1[rr];
#pragma unroll
            for (int off = 1; off < 16; off <<= 1)
#pragma unroll
                for (int rr = 0; rr < 8; ++rr)
                    ps[rr] += __shfl_xor(ps[rr], off, 32);
#pragma unroll
            for (int rr = 0; rr < 8; ++rr) lr[rr] = lr[rr] * corr[rr] + ps[rr];
#pragma unroll
            for (int t = 0; t < 8; ++t)
#pragma unroll
                for (int rr = 0; rr < 8; ++rr) o[t][rr] *= corr[rr];

            // ---- bounce P through LDS to re-layout as A fragment ----
#pragma unroll
            for (int rr = 0; rr < 8; ++rr) {
                Ps[wid][rr + 8 * hf][r]      = f2bf(p0[rr]);
                Ps[wid][rr + 8 * hf][16 + r] = f2bf(p1[rr]);
            }
            asm volatile("s_wait_dscnt 0" ::: "memory");

            v16bf pf;
            {
                const __bf16* pp = &Ps[wid][r][0];
#pragma unroll
                for (int i = 0; i < 8; ++i) {
                    pf[i]     = pp[hf * 16 + i];
                    pf[8 + i] = pp[hf * 16 + 8 + i];
                }
            }
            // ---- O += P(16x32) @ V(32xd) over 8 d-tiles ----
#pragma unroll
            for (int t = 0; t < 8; ++t) {
                const __bf16* vp2 = &Vt[t * 16 + r][(kb - c0) + hf * 16];
                v16bf b;
#pragma unroll
                for (int i = 0; i < 16; ++i) b[i] = vp2[i];
                o[t] = __builtin_amdgcn_wmma_f32_16x16x32_bf16(
                    false, pf, false, b, (short)0, o[t], false, false);
            }
        }
    }

    // finalize and store bf16 attention output [s, h*128+d]
    float inv[8];
#pragma unroll
    for (int rr = 0; rr < 8; ++rr) inv[rr] = 1.0f / lr[rr];
#pragma unroll
    for (int t = 0; t < 8; ++t)
#pragma unroll
        for (int rr = 0; rr < 8; ++rr) {
            int m = qb + rr + 8 * hf;
            O[((size_t)m * NQH + h) * HEAD_DIM + t * 16 + r] = f2bf(o[t][rr] * inv[rr]);
        }
}

// ---------------- host orchestration ----------------
extern "C" void kernel_launch(void* const* d_in, const int* in_sizes, int n_in,
                              void* d_out, int out_size, void* d_ws, size_t ws_size,
                              hipStream_t stream) {
    const float* x  = (const float*)d_in[0];
    const float* wq = (const float*)d_in[1];
    const float* wk = (const float*)d_in[2];
    const float* wv = (const float*)d_in[3];
    const float* wo = (const float*)d_in[4];
    float* out = (float*)d_out;

    // workspace carve-up (256B aligned)
    char* ws = (char*)d_ws;
    size_t off = 0;
    auto alloc = [&](size_t bytes) {
        char* p = ws + off;
        off += (bytes + 255) & ~(size_t)255;
        return p;
    };
    const size_t nx  = (size_t)SEQ * HIDDEN;            // 8.4M
    const size_t nwq = (size_t)HIDDEN * NQH * HEAD_DIM; // 16.8M
    const size_t nwk = (size_t)HIDDEN * NKVH * HEAD_DIM;// 4.2M
    const size_t nq  = (size_t)SEQ * NQH * HEAD_DIM;
    const size_t nk  = (size_t)SEQ * NKVH * HEAD_DIM;

    __bf16* xb  = (__bf16*)alloc(nx * 2);
    __bf16* wqb = (__bf16*)alloc(nwq * 2);
    __bf16* wkb = (__bf16*)alloc(nwk * 2);
    __bf16* wvb = (__bf16*)alloc(nwk * 2);
    __bf16* wob = (__bf16*)alloc(nwq * 2);
    float*  qf  = (float*) alloc(nq * 4);
    float*  kf  = (float*) alloc(nk * 4);
    __bf16* qb  = (__bf16*)alloc(nq * 2);
    __bf16* kb  = (__bf16*)alloc(nk * 2);
    __bf16* vb  = (__bf16*)alloc(nk * 2);
    __bf16* aob = (__bf16*)alloc(nq * 2);
    (void)ws_size; (void)in_sizes; (void)n_in; (void)out_size;

    auto cvt = [&](const float* src, __bf16* dst, size_t n) {
        convert_f32_bf16<<<(unsigned)((n + 255) / 256), 256, 0, stream>>>(src, dst, (int)n);
    };
    cvt(x,  xb,  nx);
    cvt(wq, wqb, nwq);
    cvt(wk, wkb, nwk);
    cvt(wv, wvb, nwk);
    cvt(wo, wob, nwq);

    // projections
    gemm_bf16_wmma<false><<<dim3((NQH * HEAD_DIM) / 128, SEQ / 128), 256, 0, stream>>>(
        xb, wqb, qf, nullptr, SEQ, NQH * HEAD_DIM, HIDDEN);
    gemm_bf16_wmma<false><<<dim3((NKVH * HEAD_DIM) / 128, SEQ / 128), 256, 0, stream>>>(
        xb, wkb, kf, nullptr, SEQ, NKVH * HEAD_DIM, HIDDEN);
    gemm_bf16_wmma<true><<<dim3((NKVH * HEAD_DIM) / 128, SEQ / 128), 256, 0, stream>>>(
        xb, wvb, nullptr, vb, SEQ, NKVH * HEAD_DIM, HIDDEN);

    // RoPE
    rope_to_bf16<<<(SEQ * NQH * 64 + 255) / 256, 256, 0, stream>>>(qf, qb, NQH);
    rope_to_bf16<<<(SEQ * NKVH * 64 + 255) / 256, 256, 0, stream>>>(kf, kb, NKVH);

    // sliding-window attention
    attn_swa_wmma<<<dim3(SEQ / 128, NQH), 256, 0, stream>>>(qb, kb, vb, aob);

    // output projection -> f32 d_out
    gemm_bf16_wmma<false><<<dim3(HIDDEN / 128, SEQ / 128), 256, 0, stream>>>(
        aob, wob, out, nullptr, SEQ, HIDDEN, NQH * HEAD_DIM);
}
